// PointNet2Encoder_320
// MI455X (gfx1250) — matrix operand order repaired and checked by reading, not compiled, not run on hardware
//
#include <hip/hip_runtime.h>

typedef _Float16 v8h  __attribute__((ext_vector_type(8)));
typedef _Float16 v16h __attribute__((ext_vector_type(16)));
typedef float    v8f  __attribute__((ext_vector_type(8)));

union V16U { v8h h[2]; v16h v; };

// ---------------------------------------------------------------------------
// Prep: fold conv-bias + BN into per-channel scale/shift; convert W to f16
// with Cin padded to a multiple of 32 (zeros) for the WMMA K-loop.
// ---------------------------------------------------------------------------
__global__ void prep_layer_kernel(const float* __restrict__ W, const float* __restrict__ bias,
                                  const float* __restrict__ gamma, const float* __restrict__ beta,
                                  const float* __restrict__ mean, const float* __restrict__ var,
                                  _Float16* __restrict__ w16, float* __restrict__ scale,
                                  float* __restrict__ shift, int cin, int cinPad, int cout) {
  int t = blockIdx.x * blockDim.x + threadIdx.x;
  if (t >= cout * cinPad) return;
  int o = t / cinPad, k = t - o * cinPad;
  w16[t] = (k < cin) ? (_Float16)W[o * cin + k] : (_Float16)0.0f;
  if (k == 0) {
    float s = gamma[o] / sqrtf(var[o] + 1e-5f);
    scale[o] = s;
    shift[o] = (bias[o] - mean[o]) * s + beta[o];
  }
}

// ---------------------------------------------------------------------------
// Farthest point sampling: one workgroup per batch, points + running min-dist
// cached in registers (<=4 per thread), block-wide argmax with exact
// lowest-index tie-breaking to match jnp.argmax.
// ---------------------------------------------------------------------------
__global__ void fps_kernel(const float* __restrict__ xyz, int N, int npoint,
                           int* __restrict__ out) {
  const int b = blockIdx.x;
  const int t = threadIdx.x;
  const float* P = xyz + (size_t)b * N * 3;
  float px[4], py[4], pz[4], dist[4];
#pragma unroll
  for (int j = 0; j < 4; j++) {
    int i = t + j * 1024;
    if (i < N) { px[j] = P[i * 3 + 0]; py[j] = P[i * 3 + 1]; pz[j] = P[i * 3 + 2]; }
    else       { px[j] = 0.f; py[j] = 0.f; pz[j] = 0.f; }
    dist[j] = 1e10f;
  }
  __shared__ float rv[1024];
  __shared__ int   ri[1024];
  __shared__ float cx, cy, cz;
  __shared__ int   sfar;
  if (t == 0) sfar = 0;
  __syncthreads();
  for (int it = 0; it < npoint; ++it) {
    int far = sfar;
    if (t == 0) {
      out[b * npoint + it] = far;
      cx = P[far * 3 + 0]; cy = P[far * 3 + 1]; cz = P[far * 3 + 2];
    }
    __syncthreads();
    float bestv = -1.f; int besti = 0;
#pragma unroll
    for (int j = 0; j < 4; j++) {
      int i = t + j * 1024;
      if (i < N) {
        float dx = px[j] - cx, dy = py[j] - cy, dz = pz[j] - cz;
        float d = dx * dx + dy * dy + dz * dz;
        float nd = fminf(dist[j], d);
        dist[j] = nd;
        if (nd > bestv || (nd == bestv && i < besti)) { bestv = nd; besti = i; }
      }
    }
    rv[t] = bestv; ri[t] = besti;
    __syncthreads();
    for (int s = 512; s > 0; s >>= 1) {
      if (t < s) {
        float v2 = rv[t + s]; int i2 = ri[t + s];
        if (v2 > rv[t] || (v2 == rv[t] && i2 < ri[t])) { rv[t] = v2; ri[t] = i2; }
      }
      __syncthreads();
    }
    if (t == 0) sfar = ri[0];
    __syncthreads();
  }
}

__global__ void gather_points_kernel(const float* __restrict__ src, const int* __restrict__ idx,
                                     float* __restrict__ dst, int N, int Snp, int total) {
  int t = blockIdx.x * blockDim.x + threadIdx.x;
  if (t >= total) return;
  int b = t / Snp;
  int i = idx[t];
  const float* p = src + ((size_t)b * N + i) * 3;
  dst[(size_t)t * 3 + 0] = p[0];
  dst[(size_t)t * 3 + 1] = p[1];
  dst[(size_t)t * 3 + 2] = p[2];
}

// ---------------------------------------------------------------------------
// Ball query: one wave32 per center; ballot + prefix popcount yields the
// first-K in-radius indices in ascending order (== reference's sort+slice),
// padded with the first hit.
// ---------------------------------------------------------------------------
__global__ void ball_query_kernel(const float* __restrict__ xyz, const float* __restrict__ centers,
                                  int N, int S, int K, float r2, int* __restrict__ out) {
  int wave = (blockIdx.x * blockDim.x + threadIdx.x) >> 5;
  int lane = threadIdx.x & 31;
  int b = wave / S, s = wave - b * S;
  const float* P = xyz + (size_t)b * N * 3;
  const float* c = centers + ((size_t)b * S + s) * 3;
  float cxx = c[0], cyy = c[1], czz = c[2];
  int* o = out + ((size_t)b * S + s) * K;
  int count = 0, first = 0;
  for (int base = 0; base < N; base += 32) {
    int i = base + lane;
    bool valid = false;
    if (i < N) {
      float dx = P[i * 3 + 0] - cxx, dy = P[i * 3 + 1] - cyy, dz = P[i * 3 + 2] - czz;
      valid = (dx * dx + dy * dy + dz * dz) <= r2;
    }
    unsigned mask = (unsigned)__ballot(valid);
    if (count == 0 && mask) first = base + __builtin_ctz(mask);
    if (valid) {
      int pos = count + __popc(mask & ((1u << lane) - 1u));
      if (pos < K) o[pos] = i;
    }
    count += __popc(mask);
    if (count >= K) break;
  }
  int fs = count < K ? count : K;
  for (int j = fs + lane; j < K; j += 32) o[j] = first;
}

// Stage-1 grouping: centered xyz deltas -> X[B, S*K, 32] f16 (channels 3..31 = 0)
__global__ void group1_kernel(const float* __restrict__ xyz, const float* __restrict__ centers,
                              const int* __restrict__ idx, _Float16* __restrict__ X,
                              int N, int S, int K, int total) {
  int t = blockIdx.x * blockDim.x + threadIdx.x;
  if (t >= total) return;
  int s = (t / K) % S;
  int b = t / (K * S);
  int i = idx[t];
  const float* p = xyz + ((size_t)b * N + i) * 3;
  const float* c = centers + ((size_t)b * S + s) * 3;
  v8h h0 = {};
  h0[0] = (_Float16)(p[0] - c[0]);
  h0[1] = (_Float16)(p[1] - c[1]);
  h0[2] = (_Float16)(p[2] - c[2]);
  v8h z = {};
  _Float16* row = X + (size_t)t * 32;
  *(v8h*)(row)      = h0;
  *(v8h*)(row + 8)  = z;
  *(v8h*)(row + 16) = z;
  *(v8h*)(row + 24) = z;
}

// Stage-2 grouping: gather feature rows (f16, 16B chunks)
__global__ void group2_kernel(const _Float16* __restrict__ feat, const int* __restrict__ idx,
                              _Float16* __restrict__ X, int Sprev, int C, int SK, int total) {
  int t = blockIdx.x * blockDim.x + threadIdx.x;
  int chunks = C >> 3;
  int pos = t / chunks, c8 = t - pos * chunks;
  if (pos >= total) return;
  int b = pos / SK;
  int i = idx[pos];
  const v8h* src = (const v8h*)(feat + ((size_t)b * Sprev + i) * C);
  v8h* dst = (v8h*)(X + (size_t)pos * C);
  dst[c8] = src[c8];
}

// ---------------------------------------------------------------------------
// Generic 1x1-conv + BN + ReLU layer via V_WMMA_F32_16X16X32_F16.
// X: [B, P, Cs] f16 (position-major), W: [Cout, Cs] f16, Y: [B, P, Cout] f16.
// One wave computes a 32(cout) x 32(pos) tile as 2x2 WMMA tiles (4 f32
// accumulators); each A/B fragment is reused by two WMMAs, halving cache
// traffic per MAC vs a single-tile wave (~16 FLOP/byte from L0/L2).
// Fragment addressing follows ISA 7.12.2 wave32 layouts:
//   A (16x32 f16): lane m = lane&15; halves K in [8*hi,8*hi+8) u [16+8*hi,+8)
//   B (32x16 f16): lane n = lane&15; halves K in [8*hi,8*hi+8) u [16+8*hi,+8) (same as A)
//   C/D (16x16 f32): VGPR r -> M = r + 8*hi, N = lane&15
// ---------------------------------------------------------------------------
__global__ void wmma_layer_kernel(const _Float16* __restrict__ X, const _Float16* __restrict__ W,
                                  const float* __restrict__ scale, const float* __restrict__ shift,
                                  _Float16* __restrict__ Y, int P, int Cs, int Cout) {
  const int lane = threadIdx.x & 31;
  const int tile = blockIdx.x * (blockDim.x >> 5) + (threadIdx.x >> 5);
  const int pt2 = P >> 5;                      // 32-wide position tiles
  const int ot2 = Cout >> 5;                   // 32-wide cout tiles
  int b = tile / (pt2 * ot2);
  int r = tile - b * pt2 * ot2;
  int om = r / pt2;
  int oc = om << 5;
  int pb = (r - om * pt2) << 5;
  const int hi = lane >> 4;
  const int lo = lane & 15;
  const _Float16* wr0 = W + (size_t)(oc + lo) * Cs;
  const _Float16* wr1 = wr0 + (size_t)16 * Cs;
  const _Float16* xr0 = X + ((size_t)b * P + pb + lo) * Cs;
  const _Float16* xr1 = xr0 + (size_t)16 * Cs;
  v8f acc00 = {0.f, 0.f, 0.f, 0.f, 0.f, 0.f, 0.f, 0.f};
  v8f acc01 = acc00, acc10 = acc00, acc11 = acc00;
  for (int k0 = 0; k0 < Cs; k0 += 32) {
    V16U A0, A1, B0, B1;
    A0.h[0] = *(const v8h*)(wr0 + k0 + 8 * hi);
    A0.h[1] = *(const v8h*)(wr0 + k0 + 16 + 8 * hi);
    A1.h[0] = *(const v8h*)(wr1 + k0 + 8 * hi);
    A1.h[1] = *(const v8h*)(wr1 + k0 + 16 + 8 * hi);
    B0.h[0] = *(const v8h*)(xr0 + k0 + 8 * hi);
    B0.h[1] = *(const v8h*)(xr0 + k0 + 16 + 8 * hi);
    B1.h[0] = *(const v8h*)(xr1 + k0 + 8 * hi);
    B1.h[1] = *(const v8h*)(xr1 + k0 + 16 + 8 * hi);
    acc00 = __builtin_amdgcn_wmma_f32_16x16x32_f16(false, A0.v, false, B0.v,
                                                   (short)0, acc00, false, false);
    acc01 = __builtin_amdgcn_wmma_f32_16x16x32_f16(false, A0.v, false, B1.v,
                                                   (short)0, acc01, false, false);
    acc10 = __builtin_amdgcn_wmma_f32_16x16x32_f16(false, A1.v, false, B0.v,
                                                   (short)0, acc10, false, false);
    acc11 = __builtin_amdgcn_wmma_f32_16x16x32_f16(false, A1.v, false, B1.v,
                                                   (short)0, acc11, false, false);
  }
#pragma unroll
  for (int mi = 0; mi < 2; ++mi) {
    int ob = oc + mi * 16 + 8 * hi;
    float4 s0 = *(const float4*)(scale + ob);
    float4 s1 = *(const float4*)(scale + ob + 4);
    float4 f0 = *(const float4*)(shift + ob);
    float4 f1 = *(const float4*)(shift + ob + 4);
    float sc[8] = {s0.x, s0.y, s0.z, s0.w, s1.x, s1.y, s1.z, s1.w};
    float sf[8] = {f0.x, f0.y, f0.z, f0.w, f1.x, f1.y, f1.z, f1.w};
#pragma unroll
    for (int ni = 0; ni < 2; ++ni) {
      v8f a = (mi == 0) ? (ni == 0 ? acc00 : acc01) : (ni == 0 ? acc10 : acc11);
      int p = pb + ni * 16 + lo;
      v8h outv;
#pragma unroll
      for (int rr = 0; rr < 8; ++rr) {
        float v = a[rr] * sc[rr] + sf[rr];
        v = fmaxf(v, 0.f);
        outv[rr] = (_Float16)v;
      }
      *(v8h*)(Y + ((size_t)b * P + p) * Cout + ob) = outv;
    }
  }
}

// Max over the k (neighbor) axis: [B, S*K, C] -> [B, S, C] (f16, exact)
__global__ void maxk_kernel(const _Float16* __restrict__ Y, _Float16* __restrict__ out,
                            int S, int K, int C, int total) {
  int t = blockIdx.x * blockDim.x + threadIdx.x;
  if (t >= total) return;
  int c = t % C;
  int s = (t / C) % S;
  int b = t / (C * S);
  const _Float16* base = Y + ((size_t)b * S + s) * (size_t)K * C + c;
  float m = -3.4e38f;
  for (int j = 0; j < K; j++) m = fmaxf(m, (float)base[(size_t)j * C]);
  out[((size_t)b * S + s) * C + c] = (_Float16)m;
}

// Final max over the 128 positions -> f32 output [B, 512]
__global__ void final_max_kernel(const _Float16* __restrict__ Y, float* __restrict__ out,
                                 int K, int C, int total) {
  int t = blockIdx.x * blockDim.x + threadIdx.x;
  if (t >= total) return;
  int c = t % C;
  int b = t / C;
  const _Float16* base = Y + (size_t)b * K * C + c;
  float m = -3.4e38f;
  for (int j = 0; j < K; j++) m = fmaxf(m, (float)base[(size_t)j * C]);
  out[t] = m;
}

// ---------------------------------------------------------------------------
static inline size_t alignUp256(size_t x) { return (x + 255) & ~(size_t)255; }

extern "C" void kernel_launch(void* const* d_in, const int* in_sizes, int n_in,
                              void* d_out, int out_size, void* d_ws, size_t ws_size,
                              hipStream_t stream) {
  const float* xyz = (const float*)d_in[0];
  const int N1 = 4096;
  const int B = in_sizes[0] / (N1 * 3);            // 16
  const int S1 = 512, K1 = 32;
  const int S2 = 128, K2 = 64;

  struct LDim { int cin, cinPad, cout; };
  const LDim ld[9] = {{3, 32, 64},   {64, 64, 64},   {64, 64, 128},
                      {128, 128, 128},{128, 128, 128},{128, 128, 256},
                      {256, 256, 256},{256, 256, 512},{512, 512, 512}};
  size_t wOff[9]; int cOff[9];
  size_t wtot = 0; int ctot = 0;
  for (int i = 0; i < 9; i++) {
    wOff[i] = wtot; wtot += (size_t)ld[i].cout * ld[i].cinPad;
    cOff[i] = ctot; ctot += ld[i].cout;
  }

  char* ws = (char*)d_ws;
  size_t off = 0;
  _Float16* w16   = (_Float16*)(ws + off); off = alignUp256(off + wtot * 2);
  float*    scl   = (float*)(ws + off);    off = alignUp256(off + (size_t)ctot * 4);
  float*    shf   = (float*)(ws + off);    off = alignUp256(off + (size_t)ctot * 4);
  int*      fps1  = (int*)(ws + off);      off = alignUp256(off + (size_t)B * S1 * 4);
  float*    nxyz1 = (float*)(ws + off);    off = alignUp256(off + (size_t)B * S1 * 3 * 4);
  int*      ball1 = (int*)(ws + off);      off = alignUp256(off + (size_t)B * S1 * K1 * 4);
  int*      fps2  = (int*)(ws + off);      off = alignUp256(off + (size_t)B * S2 * 4);
  float*    nxyz2 = (float*)(ws + off);    off = alignUp256(off + (size_t)B * S2 * 3 * 4);
  int*      ball2 = (int*)(ws + off);      off = alignUp256(off + (size_t)B * S2 * K2 * 4);
  _Float16* feat1 = (_Float16*)(ws + off); off = alignUp256(off + (size_t)B * S1 * 128 * 2);
  _Float16* feat2 = (_Float16*)(ws + off); off = alignUp256(off + (size_t)B * S2 * 256 * 2);
  _Float16* buf0  = (_Float16*)(ws + off); off = alignUp256(off + (size_t)B * 16384 * 64 * 2);
  _Float16* buf1  = (_Float16*)(ws + off); off = alignUp256(off + (size_t)B * 16384 * 128 * 2);
  (void)ws_size; (void)n_in; (void)out_size;

  // --- fold params ---
  for (int i = 0; i < 9; i++) {
    const float* W  = (const float*)d_in[1 + 6 * i + 0];
    const float* bb = (const float*)d_in[1 + 6 * i + 1];
    const float* ga = (const float*)d_in[1 + 6 * i + 2];
    const float* be = (const float*)d_in[1 + 6 * i + 3];
    const float* me = (const float*)d_in[1 + 6 * i + 4];
    const float* va = (const float*)d_in[1 + 6 * i + 5];
    int th = ld[i].cout * ld[i].cinPad;
    prep_layer_kernel<<<(th + 255) / 256, 256, 0, stream>>>(
        W, bb, ga, be, me, va, w16 + wOff[i], scl + cOff[i], shf + cOff[i],
        ld[i].cin, ld[i].cinPad, ld[i].cout);
  }

  auto layer = [&](int li, const _Float16* X, _Float16* Y, int P) {
    int tiles = B * (P / 32) * (ld[li].cout / 32);   // always a multiple of 8
    wmma_layer_kernel<<<tiles / 8, 256, 0, stream>>>(
        X, w16 + wOff[li], scl + cOff[li], shf + cOff[li], Y, P, ld[li].cinPad, ld[li].cout);
  };

  // ---------------- Stage 1 ----------------
  fps_kernel<<<B, 1024, 0, stream>>>(xyz, N1, S1, fps1);
  {
    int th = B * S1;
    gather_points_kernel<<<(th + 255) / 256, 256, 0, stream>>>(xyz, fps1, nxyz1, N1, S1, th);
  }
  ball_query_kernel<<<(B * S1) / 8, 256, 0, stream>>>(xyz, nxyz1, N1, S1, K1, 0.2f * 0.2f, ball1);
  {
    int th = B * S1 * K1;
    group1_kernel<<<(th + 255) / 256, 256, 0, stream>>>(xyz, nxyz1, ball1, buf0, N1, S1, K1, th);
  }
  const int P1 = S1 * K1;                 // 16384
  layer(0, buf0, buf1, P1);               // 32 -> 64
  layer(1, buf1, buf0, P1);               // 64 -> 64
  layer(2, buf0, buf1, P1);               // 64 -> 128
  {
    int th = B * S1 * 128;
    maxk_kernel<<<(th + 255) / 256, 256, 0, stream>>>(buf1, feat1, S1, K1, 128, th);
  }

  // ---------------- Stage 2 ----------------
  fps_kernel<<<B, 1024, 0, stream>>>(nxyz1, S1, S2, fps2);
  {
    int th = B * S2;
    gather_points_kernel<<<(th + 255) / 256, 256, 0, stream>>>(nxyz1, fps2, nxyz2, S1, S2, th);
  }
  ball_query_kernel<<<(B * S2) / 8, 256, 0, stream>>>(nxyz1, nxyz2, S1, S2, K2, 0.4f * 0.4f, ball2);
  {
    int positions = B * S2 * K2;
    int th = positions * (128 / 8);
    group2_kernel<<<(th + 255) / 256, 256, 0, stream>>>(feat1, ball2, buf0, S1, 128, S2 * K2, positions);
  }
  const int P2 = S2 * K2;                 // 8192
  layer(3, buf0, buf1, P2);               // 128 -> 128
  layer(4, buf1, buf0, P2);               // 128 -> 128
  layer(5, buf0, buf1, P2);               // 128 -> 256
  {
    int th = B * S2 * 256;
    maxk_kernel<<<(th + 255) / 256, 256, 0, stream>>>(buf1, feat2, S2, K2, 256, th);
  }

  // ---------------- Stage 3 ----------------
  const int P3 = 128;                     // k=128 positions, S=1
  layer(6, feat2, buf0, P3);              // 256 -> 256
  layer(7, buf0, buf1, P3);               // 256 -> 512
  layer(8, buf1, buf0, P3);               // 512 -> 512
  {
    int th = B * 512;
    final_max_kernel<<<(th + 255) / 256, 256, 0, stream>>>(buf0, (float*)d_out, P3, 512, th);
  }
}
